// Decoder_88046829568673
// MI455X (gfx1250) — matrix operand order repaired and checked by reading, not compiled, not run on hardware
//
#include <hip/hip_runtime.h>

// ---------------- model dims ----------------
#define Bn     32
#define NMEL   80
#define MEMD   512
#define PN     256
#define AL     1024
#define AD     128
#define NS     8
#define SFK    21
#define NDY    8
#define DFK    21
#define PL     11
#define DL     1024
#define RR     2
#define TM     800
#define TE     200
#define NSTEP  400
#define NG     4096          // 4*1024 gates
#define KA     1792          // ctx(512)+prenet(256)+h(1024)
#define KD     2560          // h_attn(1024)+ctx(512)+h_dec(1024)
#define KTA    (KA/32)       // 56 k-tiles
#define KTD    (KD/32)       // 80 k-tiles
#define NTIL   (NG/16)       // 256 n-tiles
#define NWG    32            // one block per batch row; 32 WGPs pulling L2
#define TPB    512           // 16 waves/WG -> 4 waves/SIMD
#define PBK    256           // block size for pack/prenet kernels
#define NWAVES (NWG*TPB/32)  // 512 == number of 16x16 output tiles per GEMM
#define CH     8             // k-tiles staged per LDS chunk (KTA=7*8, KTD=10*8)
#define CHUSH  (CH*1024)     // ushorts per chunk: CH * 2 mt * 32 lanes * 16

static_assert(NWAVES == 2 * NTIL, "one 16x16 tile per wave");
static_assert(KTA % CH == 0 && KTD % CH == 0, "chunked k-tiling");

typedef __bf16 v16bf __attribute__((ext_vector_type(16)));
typedef float  v8f   __attribute__((ext_vector_type(8)));

__device__ __forceinline__ unsigned short f2bf(float f) {
  unsigned u = __float_as_uint(f);
  u += 0x7fffu + ((u >> 16) & 1u);        // round-to-nearest-even
  return (unsigned short)(u >> 16);
}
__device__ __forceinline__ float sigm(float x) { return 1.f / (1.f + __expf(-x)); }

// ushort index of element k of batch-row b inside the A-fragment-layout buffer
// Layout: [kt][mt][lane][16 bf16], lane = half*16 + (b&15), per ISA A 16x32 map.
__device__ __forceinline__ size_t fragIdx(int b, int k) {
  const int kt = k >> 5, kk = k & 31;
  const int half = (kk >> 3) & 1;
  const int q = ((kk >> 4) ? 4 : 0) + ((kk & 7) >> 1);
  const int i = kk & 1;
  const int mt = b >> 4, lane = half * 16 + (b & 15);
  return ((size_t)(kt * 2 + mt) * 32 + lane) * 16 + q * 2 + i;
}

// -------- grid-wide barrier (32 co-resident WGs, sense-reversal) -----------
__device__ __forceinline__ void gbar(unsigned* bar) {
  __threadfence();
  __syncthreads();
  if (threadIdx.x == 0) {
    volatile unsigned* vgen = bar + 1;
    unsigned gen = *vgen;
    if (atomicAdd(bar, 1u) == NWG - 1) {
      *((volatile unsigned*)bar) = 0u;
      __threadfence();
      atomicAdd(bar + 1, 1u);
    } else {
      while (*vgen == gen) { __builtin_amdgcn_s_sleep(2); }
    }
  }
  __syncthreads();
  __threadfence();
}

// -------- bf16 WMMA GEMM: gates[32][4096] = X[32][K] * Wpacked^T -----------
// A staged through double-buffered LDS chunks (shared by all 16 waves of the
// WG -> A read from L2 once per WG, and A sits on DScnt while B streams on
// LOADcnt with deep prefetch). B: fragment-major packed, 32B/lane contiguous.
__device__ __forceinline__ void gemm_tiles(const unsigned short* __restrict__ xf,
                                           int ktiles,
                                           const unsigned short* __restrict__ wp,
                                           float* __restrict__ gates,
                                           unsigned short* l0, unsigned short* l1,
                                           int wave, int lane, int tid) {
  const int m16 = lane & 15, half = lane >> 4;
  const int mt = wave & 1, nt = wave >> 1;          // one 16x16 tile per wave
  const unsigned short* wt = wp + (size_t)nt * ktiles * 512 + lane * 16;
  v8f acc = {};
  const int nch = ktiles / CH;
  { // stage chunk 0
    const uint4* src = (const uint4*)xf;
    uint4* dst = (uint4*)l0;
    for (int i = tid; i < CHUSH / 8; i += TPB) dst[i] = src[i];
  }
  __syncthreads();
  for (int c = 0; c < nch; ++c) {
    unsigned short* cur = (c & 1) ? l1 : l0;
    unsigned short* nxt = (c & 1) ? l0 : l1;
    if (c + 1 < nch) {  // prefetch next chunk while computing this one
      const uint4* src = (const uint4*)(xf + (size_t)(c + 1) * CHUSH);
      uint4* dst = (uint4*)nxt;
      for (int i = tid; i < CHUSH / 8; i += TPB) dst[i] = src[i];
    }
    const unsigned short* bt = wt + (size_t)c * CH * 512;
#pragma unroll
    for (int k = 0; k < CH; ++k) {
      v16bf a = *reinterpret_cast<const v16bf*>(cur + ((size_t)(k * 2 + mt) * 32 + lane) * 16);
      v16bf bfr = *reinterpret_cast<const v16bf*>(bt + (size_t)k * 512);
      acc = __builtin_amdgcn_wmma_f32_16x16x32_bf16(false, a, false, bfr,
                                                    (short)0, acc, false, false);
    }
    __syncthreads();  // all waves done with 'cur' before it is restaged
  }
  float* gp = gates + (size_t)(mt * 16 + half * 8) * NG + nt * 16 + m16;
#pragma unroll
  for (int r = 0; r < 8; ++r) gp[(size_t)r * NG] = acc[r];
}

// -------- pack fp32 weights -> bf16 WMMA-B fragment-major ------------------
__global__ void k_pack(const float* __restrict__ aWih, const float* __restrict__ aWhh,
                       const float* __restrict__ dWih, const float* __restrict__ dWhh,
                       unsigned short* __restrict__ wa, unsigned short* __restrict__ wd,
                       unsigned* bar) {
  const size_t stride = (size_t)gridDim.x * blockDim.x;
  const size_t id0 = (size_t)blockIdx.x * blockDim.x + threadIdx.x;
  const size_t PA = (size_t)NTIL * KTA * 512;
  for (size_t p = id0; p < PA; p += stride) {
    int r = (int)(p & 511); int tile = (int)(p >> 9);
    int kt = tile % KTA, ntile = tile / KTA;
    int lane = r >> 4, i = r & 15;
    int n = ntile * 16 + (lane & 15);
    int k = kt * 32 + (i >> 3) * 16 + (lane >> 4) * 8 + (i & 7);
    float w = (k < 768) ? aWih[(size_t)n * 768 + k] : aWhh[(size_t)n * 1024 + (k - 768)];
    wa[p] = f2bf(w);
  }
  const size_t PD = (size_t)NTIL * KTD * 512;
  for (size_t p = id0; p < PD; p += stride) {
    int r = (int)(p & 511); int tile = (int)(p >> 9);
    int kt = tile % KTD, ntile = tile / KTD;
    int lane = r >> 4, i = r & 15;
    int n = ntile * 16 + (lane & 15);
    int k = kt * 32 + (i >> 3) * 16 + (lane >> 4) * 8 + (i & 7);
    float w = (k < 1536) ? dWih[(size_t)n * 1536 + k] : dWhh[(size_t)n * 1024 + (k - 1536)];
    wd[p] = f2bf(w);
  }
  if (id0 == 0) { bar[0] = 0u; bar[1] = 0u; }
}

// -------- hoisted prenet (teacher forcing => not on recurrent path) --------
__global__ void k_prenet(const float* __restrict__ mels, const float* __restrict__ W1,
                         const float* __restrict__ b1, const float* __restrict__ W2,
                         const float* __restrict__ b2, unsigned short* __restrict__ x2) {
  __shared__ float x1[PN];
  const int sb = blockIdx.x;            // 400*32 blocks
  const int s = sb >> 5, b = sb & 31;
  const int j = threadIdx.x;            // 256
  float z = b1[j];
  if (s > 0) {
    const float* w = W1 + (size_t)j * NMEL;
    const int t = 2 * s - 1;            // mels[:, :, R-1::R][:, :, s-1]
#pragma unroll 4
    for (int m = 0; m < NMEL; ++m) z += mels[((size_t)b * NMEL + m) * TM + t] * w[m];
  }
  x1[j] = fmaxf(z, 0.f);
  __syncthreads();
  float z2 = b2[j];
  const float* w2 = W2 + (size_t)j * PN;
#pragma unroll 4
  for (int k = 0; k < PN; ++k) z2 += x1[k] * w2[k];
  x2[((size_t)s * Bn + b) * PN + j] = f2bf(fmaxf(z2, 0.f));
}

struct DecP {
  unsigned* bar;
  const unsigned short* wa; const unsigned short* wd; const unsigned short* x2;
  unsigned short* xa; unsigned short* xd;   // A-fragment-layout activation buffers
  float* gates;
  const float* abih; const float* abhh; const float* dbih; const float* dbhh;
  const float* wq; const float* bq; const float* V; const float* F;
  const float* U; const float* Tw; const float* Tb; const float* vv; const float* prior;
  const float* mem; const float* acW; const float* acb;
  float* out;
};

// -------- persistent decoder ------------------------------------------------
// Block b permanently owns batch row b: all recurrent state lives in LDS for
// the whole 400-step run. Only WMMA inputs (xa/xd, fragment layout) and GEMM
// outputs (gates) round-trip through L2. 4 grid barriers per step.
__global__ void __launch_bounds__(TPB, 1) k_decode(DecP P) {
  __shared__ __align__(32) unsigned short stA[2][CHUSH];  // 2 x 16 KB A stage
  __shared__ float hs[AL];   // attn LSTM h
  __shared__ float ca[AL];   // attn LSTM c
  __shared__ float hd[DL];   // dec LSTM h
  __shared__ float cd[DL];   // dec LSTM c
  __shared__ float cs[MEMD]; // context
  __shared__ float aw[TE];   // attention weights
  __shared__ float es[TE];   // energies
  __shared__ float red[TPB]; // reductions
  __shared__ float qs[AD];   // tanh(Wq h + bq)
  __shared__ float gs[NDY * DFK]; // dynamic filters

  const int tid = threadIdx.x;
  const int b = blockIdx.x;                 // NWG == Bn
  const int wave = (b * TPB + tid) >> 5, lane = tid & 31;
  float* outA = P.out + (size_t)Bn * NMEL * TM;   // alignments after mels

  // ---- init state + xa for step 0 ----
  for (int j = tid; j < AL; j += TPB) { hs[j] = 0.f; ca[j] = 0.f; hd[j] = 0.f; cd[j] = 0.f; }
  for (int t = tid; t < TE; t += TPB) aw[t] = (t == 0) ? 1.f : 0.f;
  for (int k = tid; k < KA; k += TPB)
    P.xa[fragIdx(b, k)] = (k >= 512 && k < 768) ? P.x2[(size_t)b * PN + (k - 512)]
                                                : (unsigned short)0;
  gbar(P.bar);

  for (int s = 0; s < NSTEP; ++s) {
    // ===== PH1: attn gates = xa * [Wih|Whh]^T  (WMMA bf16, all blocks) =====
    gemm_tiles(P.xa, KTA, P.wa, P.gates, stA[0], stA[1], wave, lane, tid);
    gbar(P.bar);

    // ===== PH2: attn cell + full DCA + build xd and next xa (row b only) ===
    {
      const float* g = P.gates + (size_t)b * NG;
      for (int j = tid; j < AL; j += TPB) {
        float gi = g[j]          + P.abih[j]          + P.abhh[j];
        float gf = g[AL + j]     + P.abih[AL + j]     + P.abhh[AL + j];
        float gg = g[2 * AL + j] + P.abih[2 * AL + j] + P.abhh[2 * AL + j];
        float go = g[3 * AL + j] + P.abih[3 * AL + j] + P.abhh[3 * AL + j];
        float c = sigm(gf) * ca[j] + sigm(gi) * tanhf(gg);
        ca[j] = c;
        hs[j] = sigm(go) * tanhf(c);
      }
      __syncthreads();
      // q2 = tanh(h @ Wq^T + bq): 4 threads per output (K split 4 ways)
      {
        int o = tid >> 2, part = tid & 3;
        const float* w = P.wq + (size_t)o * AL + part * 256;
        const float* h = hs + part * 256;
        float z = 0.f;
        for (int k = 0; k < 256; ++k) z += h[k] * w[k];
        red[tid] = z;
      }
      __syncthreads();
      if ((tid & 3) == 0) {
        int o = tid >> 2;
        qs[o] = tanhf(red[tid] + red[tid + 1] + red[tid + 2] + red[tid + 3] + P.bq[o]);
      }
      __syncthreads();
      // G = q2 @ V^T  [168]
      if (tid < NDY * DFK) {
        const float* v = P.V + (size_t)tid * AD;
        float z = 0.f;
        for (int k = 0; k < AD; ++k) z += qs[k] * v[k];
        gs[tid] = z;
      }
      __syncthreads();
      // e[t] = v.tanh(U f + T g + Tb) + log(prior-conv); 2 threads per t
      if (tid < 2 * TE) {
        const int t = tid >> 1, part = tid & 1;
        float win[SFK];
#pragma unroll
        for (int j = 0; j < SFK; ++j) { int tt = t + j - 10; win[j] = (tt >= 0 && tt < TE) ? aw[tt] : 0.f; }
        float fs[NS];
#pragma unroll
        for (int c = 0; c < NS; ++c) { float z = 0.f; for (int j = 0; j < SFK; ++j) z += win[j] * P.F[c * SFK + j]; fs[c] = z; }
        float gd[NDY];
#pragma unroll
        for (int d = 0; d < NDY; ++d) { float z = 0.f; const float* gk = gs + d * DFK; for (int j = 0; j < DFK; ++j) z += win[j] * gk[j]; gd[d] = z; }
        float e = 0.f;
        const int a0 = part * (AD / 2);
        for (int a = a0; a < a0 + AD / 2; ++a) {
          float z = P.Tb[a];
          const float* Ua = P.U + a * NS;
          const float* Ta = P.Tw + a * NDY;
#pragma unroll
          for (int c = 0; c < NS; ++c) z += Ua[c] * fs[c];
#pragma unroll
          for (int d = 0; d < NDY; ++d) z += Ta[d] * gd[d];
          e += P.vv[a] * tanhf(z);
        }
        if (part == 0) {
          float p = 0.f;
#pragma unroll
          for (int j = 0; j < PL; ++j) p += win[j] * P.prior[j];   // causal offsets t-10..t
          e += logf(fmaxf(p, 1e-6f));
        }
        red[tid] = e;
      }
      __syncthreads();
      if (tid < TE) es[tid] = red[2 * tid] + red[2 * tid + 1];
      __syncthreads();
      // softmax over t (block tree reduction)
      red[tid] = (tid < TE) ? es[tid] : -3.0e38f;
      __syncthreads();
      for (int o = TPB / 2; o; o >>= 1) { if (tid < o) red[tid] = fmaxf(red[tid], red[tid + o]); __syncthreads(); }
      float mx = red[0];
      __syncthreads();
      red[tid] = (tid < TE) ? __expf(es[tid] - mx) : 0.f;
      __syncthreads();
      for (int o = TPB / 2; o; o >>= 1) { if (tid < o) red[tid] += red[tid + o]; __syncthreads(); }
      float inv = 1.f / red[0];
      __syncthreads();
      if (tid < TE) {
        float a = __expf(es[tid] - mx) * inv;
        aw[tid] = a;
        outA[((size_t)b * TE + tid) * NSTEP + s] = a;
      }
      __syncthreads();
      // ctx = aw.memory (one output per thread); fragment-layout writes
      {
        const int d = tid;                    // TPB == MEMD
        const float* mem = P.mem + ((size_t)b * TE) * MEMD + d;
        float z = 0.f;
        for (int t = 0; t < TE; ++t) z += aw[t] * mem[(size_t)t * MEMD];
        cs[d] = z;
        unsigned short bv = f2bf(z);
        P.xd[fragIdx(b, 1024 + d)] = bv;
        P.xa[fragIdx(b, d)] = bv;
      }
      // h parts of xd (h_attn new, h_dec previous) and next xa (h_attn new)
      for (int k = tid; k < AL; k += TPB) {
        unsigned short hv = f2bf(hs[k]);
        P.xd[fragIdx(b, k)] = hv;
        P.xa[fragIdx(b, 768 + k)] = hv;
        P.xd[fragIdx(b, 1536 + k)] = f2bf(hd[k]);
      }
      // prenet part of next xa
      if (s + 1 < NSTEP)
        for (int j = tid; j < PN; j += TPB)
          P.xa[fragIdx(b, 512 + j)] = P.x2[((size_t)(s + 1) * Bn + b) * PN + j];
    }
    gbar(P.bar);

    // ===== PH3: dec gates = xd * [Wih|Whh]^T  (WMMA bf16, all blocks) ======
    gemm_tiles(P.xd, KTD, P.wd, P.gates, stA[0], stA[1], wave, lane, tid);
    gbar(P.bar);

    // ===== PH4: dec cell + output projection (row b only) ==================
    {
      const float* g = P.gates + (size_t)b * NG;
      for (int j = tid; j < DL; j += TPB) {
        float gi = g[j]          + P.dbih[j]          + P.dbhh[j];
        float gf = g[DL + j]     + P.dbih[DL + j]     + P.dbhh[DL + j];
        float gg = g[2 * DL + j] + P.dbih[2 * DL + j] + P.dbhh[2 * DL + j];
        float go = g[3 * DL + j] + P.dbih[3 * DL + j] + P.dbhh[3 * DL + j];
        float c = sigm(gf) * cd[j] + sigm(gi) * tanhf(gg);
        cd[j] = c;
        hd[j] = sigm(go) * tanhf(c);
      }
      __syncthreads();
      // out[n] = acb[n] + [hd|cs].acW[n]  -- 2 threads per output (K split)
      if (tid < 2 * NMEL * RR) {
        const int n = tid >> 1, part = tid & 1;
        const float* w = P.acW + (size_t)n * (DL + MEMD);
        float z = 0.f;
        if (part == 0) {
          for (int k = 0; k < 768; ++k) z += hd[k] * w[k];
          z += P.acb[n];
        } else {
          for (int k = 768; k < DL; ++k) z += hd[k] * w[k];
          for (int d = 0; d < MEMD; ++d) z += cs[d] * w[DL + d];
        }
        red[tid] = z;
      }
      __syncthreads();
      if (tid < NMEL * RR) {
        float z = red[2 * tid] + red[2 * tid + 1];
        int mel = tid >> 1, r = tid & 1;
        P.out[((size_t)b * NMEL + mel) * TM + s * RR + r] = z;
      }
    }
    gbar(P.bar);
  }
}

extern "C" void kernel_launch(void* const* d_in, const int* in_sizes, int n_in,
                              void* d_out, int out_size, void* d_ws, size_t ws_size,
                              hipStream_t stream) {
  (void)in_sizes; (void)n_in; (void)out_size; (void)ws_size;
  const float* mels   = (const float*)d_in[0];
  const float* memory = (const float*)d_in[1];
  const float* pW1 = (const float*)d_in[2];  const float* pb1 = (const float*)d_in[3];
  const float* pW2 = (const float*)d_in[4];  const float* pb2 = (const float*)d_in[5];
  const float* aWih = (const float*)d_in[6]; const float* aWhh = (const float*)d_in[7];
  const float* abih = (const float*)d_in[8]; const float* abhh = (const float*)d_in[9];
  const float* wq = (const float*)d_in[10];  const float* bq = (const float*)d_in[11];
  const float* dV = (const float*)d_in[12];  const float* dF = (const float*)d_in[13];
  const float* dU = (const float*)d_in[14];  const float* dT = (const float*)d_in[15];
  const float* dTb = (const float*)d_in[16]; const float* dv = (const float*)d_in[17];
  const float* prior = (const float*)d_in[18];
  const float* dWih = (const float*)d_in[19]; const float* dWhh = (const float*)d_in[20];
  const float* dbih = (const float*)d_in[21]; const float* dbhh = (const float*)d_in[22];
  const float* acW = (const float*)d_in[23];  const float* acb = (const float*)d_in[24];

  char* ws = (char*)d_ws;
  size_t off = 0;
  auto take = [&](size_t bytes) -> void* {
    void* p = ws + off;
    off += (bytes + 255) & ~(size_t)255;
    return p;
  };
  unsigned* bar        = (unsigned*)take(256);
  unsigned short* wa   = (unsigned short*)take((size_t)NTIL * KTA * 512 * 2);
  unsigned short* wd   = (unsigned short*)take((size_t)NTIL * KTD * 512 * 2);
  unsigned short* x2   = (unsigned short*)take((size_t)NSTEP * Bn * PN * 2);
  float* gates         = (float*)take((size_t)Bn * NG * 4);
  unsigned short* xa   = (unsigned short*)take((size_t)Bn * KA * 2);
  unsigned short* xd   = (unsigned short*)take((size_t)Bn * KD * 2);

  k_pack<<<2048, PBK, 0, stream>>>(aWih, aWhh, dWih, dWhh, wa, wd, bar);
  k_prenet<<<NSTEP * Bn, PBK, 0, stream>>>(mels, pW1, pb1, pW2, pb2, x2);

  DecP P;
  P.bar = bar; P.wa = wa; P.wd = wd; P.x2 = x2; P.xa = xa; P.xd = xd;
  P.gates = gates;
  P.abih = abih; P.abhh = abhh; P.dbih = dbih; P.dbhh = dbhh;
  P.wq = wq; P.bq = bq; P.V = dV; P.F = dF; P.U = dU; P.Tw = dT; P.Tb = dTb; P.vv = dv;
  P.prior = prior; P.mem = memory; P.acW = acW; P.acb = acb;
  P.out = (float*)d_out;
  k_decode<<<NWG, TPB, 0, stream>>>(P);
}
